// Attention_2774548873273
// MI455X (gfx1250) — matrix operand order repaired and checked by reading, not compiled, not run on hardware
//
#include <hip/hip_runtime.h>

// ---------------------------------------------------------------------------
// CDNA5 (gfx1250) causal multi-head attention forward, bf16 WMMA pipeline.
//   k1: f32 -> bf16 converts (x, W_K|W_Q|W_V concat, W_O)
//   k2: QKV GEMM  [8192,1024] x [3072,1024]^T -> K,Q ([BH,S,64]) and V^T ([BH,64,S])
//       A/B tiles staged to LDS with GLOBAL_LOAD_ASYNC_TO_LDS_B128 (ASYNCcnt)
//   k3: attention per (b,head): 2-pass (stats, then z = P^T V), WMMA everywhere
//   k4: output GEMM [8192,1024] x [1024,1024]^T -> f32 out (same async staging)
// ---------------------------------------------------------------------------

#define DEV static __device__ __forceinline__

typedef __bf16 bf16_t;
typedef __attribute__((ext_vector_type(16))) __bf16 v16bf;
typedef __attribute__((ext_vector_type(8)))  __bf16 v8bf;
typedef __attribute__((ext_vector_type(8)))  float  v8f;

static constexpr int SEQ   = 1024;
static constexpr int EMB   = 1024;
static constexpr int HEADS = 16;
static constexpr int DH    = 64;
static constexpr int BATCH = 8;
static constexpr int ROWS  = BATCH * SEQ;   // 8192
static constexpr int NQKV  = 3 * EMB;       // 3072

DEV v8f wmma_bf16(v16bf a, v16bf b, v8f c) {
  // 8-arg form: (neg_a, A, neg_b, B, c_mod, C, reuse_a, reuse_b)
  return __builtin_amdgcn_wmma_f32_16x16x32_bf16(false, a, false, b, (short)0, c,
                                                 false, false);
}

DEV v8f zero8() {
  v8f z;
#pragma unroll
  for (int i = 0; i < 8; ++i) z[i] = 0.0f;
  return z;
}

// Async DMA: 16 bytes global -> LDS per lane, tracked with ASYNCcnt.
// LDS destination offset comes from a VGPR; low 32 bits of a generic __shared__ pointer
// are the LDS byte offset.
DEV void async_b128_to_lds(void* lds_dst, const void* gsrc) {
  uint32_t lds_off = (uint32_t)(uintptr_t)lds_dst;
  uint64_t addr    = (uint64_t)(uintptr_t)gsrc;
  asm volatile("global_load_async_to_lds_b128 %0, %1, off"
               :: "v"(lds_off), "v"(addr) : "memory");
}

DEV void wait_asynccnt0() {
#if __has_builtin(__builtin_amdgcn_s_wait_asynccnt)
  __builtin_amdgcn_s_wait_asynccnt(0);
#else
  asm volatile("s_wait_asynccnt 0" ::: "memory");
#endif
}

// A fragment (16x32, 16-bit): lane&15 = row M; VGPRs 0-3 hold K=(0..7)+8*laneHi,
// VGPRs 4-7 hold K=(16..23)+8*laneHi  (ISA 7.12.2, 16-bit A 16x32).
DEV v16bf load_a(const bf16_t* p, int ld, int row0, int k0) {
  const int lane = threadIdx.x & 31;
  const int hi   = lane >> 4;
  const bf16_t* r = p + (size_t)(row0 + (lane & 15)) * ld + k0 + hi * 8;
  v8bf lo = *(const v8bf*)(r);
  v8bf up = *(const v8bf*)(r + 16);
  v16bf o;
#pragma unroll
  for (int i = 0; i < 8; ++i) { o[i] = lo[i]; o[i + 8] = up[i]; }
  return o;
}

// B fragment (32x16, 16-bit): lane&15 = column N; elements 0-7 hold K=(0..7)+8*laneHi,
// elements 8-15 hold K=(16..23)+8*laneHi (same K order as load_a).
// Source must be row-major with rows indexed by N (W rows / Q rows / V^T rows).
DEV v16bf load_b(const bf16_t* p, int ld, int n0, int k0) {
  const int lane = threadIdx.x & 31;
  const int hi   = lane >> 4;
  const bf16_t* r = p + (size_t)(n0 + (lane & 15)) * ld + k0 + hi * 8;
  v8bf lo = *(const v8bf*)(r);
  v8bf up = *(const v8bf*)(r + 16);
  v16bf o;
#pragma unroll
  for (int i = 0; i < 8; ++i) { o[i] = lo[i]; o[i + 8] = up[i]; }
  return o;
}

// ---------------------------------------------------------------------------
__global__ __launch_bounds__(256) void cvt_f32_bf16(const float* __restrict__ s,
                                                    bf16_t* __restrict__ d, int n) {
  int i = blockIdx.x * blockDim.x + threadIdx.x;
  if (i < n) d[i] = (bf16_t)s[i];
}

// ---------------------------------------------------------------------------
// Shared GEMM body: 128x128 block tile, 8 waves as 4(M)x2(N), wave tile 32x64.
// Per k-step the 128x32 A and B slices are DMA'd into LDS (async-to-LDS),
// then fragments are fed from LDS (ds_load_b128). acc layout: [mi][nj] v8f.
struct GemmAcc { v8f a[2][4]; };

DEV void gemm_block_bf16(const bf16_t* __restrict__ A, const bf16_t* __restrict__ B,
                         bf16_t* sA, bf16_t* sB, int rowB, int colB, GemmAcc& acc) {
  const int wave = threadIdx.x >> 5;
  const int t    = threadIdx.x;
  const int mloc = (wave & 3) * 32;   // wave M offset in tile
  const int nloc = (wave >> 2) * 64;  // wave N offset in tile

#pragma unroll
  for (int i = 0; i < 2; ++i)
#pragma unroll
    for (int j = 0; j < 4; ++j) acc.a[i][j] = zero8();

  for (int k0 = 0; k0 < EMB; k0 += 32) {
    __syncthreads();  // previous tile fully consumed
#pragma unroll
    for (int q = 0; q < 2; ++q) {
      const int lin = q * 256 + t;
      const int row = lin >> 2;          // 0..127
      const int seg = (lin & 3) * 8;     // element offset within 32-wide k slice
      async_b128_to_lds(&sA[row * 32 + seg], A + (size_t)(rowB + row) * EMB + k0 + seg);
      async_b128_to_lds(&sB[row * 32 + seg], B + (size_t)(colB + row) * EMB + k0 + seg);
    }
    wait_asynccnt0();
    __syncthreads();  // tile visible to all waves

    v16bf a0 = load_a(sA, 32, mloc, 0);
    v16bf a1 = load_a(sA, 32, mloc + 16, 0);
    v16bf b0 = load_b(sB, 32, nloc +  0, 0);
    v16bf b1 = load_b(sB, 32, nloc + 16, 0);
    v16bf b2 = load_b(sB, 32, nloc + 32, 0);
    v16bf b3 = load_b(sB, 32, nloc + 48, 0);
    acc.a[0][0] = wmma_bf16(a0, b0, acc.a[0][0]);
    acc.a[0][1] = wmma_bf16(a0, b1, acc.a[0][1]);
    acc.a[0][2] = wmma_bf16(a0, b2, acc.a[0][2]);
    acc.a[0][3] = wmma_bf16(a0, b3, acc.a[0][3]);
    acc.a[1][0] = wmma_bf16(a1, b0, acc.a[1][0]);
    acc.a[1][1] = wmma_bf16(a1, b1, acc.a[1][1]);
    acc.a[1][2] = wmma_bf16(a1, b2, acc.a[1][2]);
    acc.a[1][3] = wmma_bf16(a1, b3, acc.a[1][3]);
  }
}

// ---------------------------------------------------------------------------
// QKV projection: C[8192,3072] = Xbf16 @ Wqkv^T, scattered into K/Q (row-major
// [BH,S,64]) and V transposed ([BH,64,S]) so attention B-operands load contiguous.
__global__ __launch_bounds__(256) void gemm_qkv_kernel(const bf16_t* __restrict__ X,
                                                       const bf16_t* __restrict__ W,
                                                       bf16_t* __restrict__ Kb,
                                                       bf16_t* __restrict__ Qb,
                                                       bf16_t* __restrict__ VT) {
  __shared__ bf16_t sA[128 * 32];
  __shared__ bf16_t sB[128 * 32];
  const int lane = threadIdx.x & 31;
  const int wave = threadIdx.x >> 5;
  const int hi = lane >> 4, nn = lane & 15;
  const int rowB = blockIdx.x * 128;
  const int colB = blockIdx.y * 128;
  const int row0 = rowB + (wave & 3) * 32;
  const int col0 = colB + (wave >> 2) * 64;

  GemmAcc acc;
  gemm_block_bf16(X, W, sA, sB, rowB, colB, acc);

#pragma unroll
  for (int i = 0; i < 2; ++i)
#pragma unroll
    for (int j = 0; j < 4; ++j)
#pragma unroll
      for (int e = 0; e < 8; ++e) {
        const int rg = row0 + i * 16 + e + 8 * hi;  // x row  (b,c)
        const int cg = col0 + j * 16 + nn;          // feature (t,a,h)
        const int tt = cg >> 10;
        const int rem = cg & 1023;
        const int a = rem >> 6, h = rem & 63;
        const int bb = rg >> 10, c = rg & 1023;
        const int bh = bb * HEADS + a;
        const bf16_t v = (bf16_t)acc.a[i][j][e];
        if (tt == 0)      Kb[((size_t)bh * SEQ + c) * DH + h] = v;
        else if (tt == 1) Qb[((size_t)bh * SEQ + c) * DH + h] = v;
        else              VT[((size_t)bh * DH + h) * SEQ + c] = v;
      }
}

// ---------------------------------------------------------------------------
// Attention: one workgroup per (b, head).  scores[c,C] = K[c]·Q[C], softmax over
// C<=c per row c (exactly reference: masked value -1e10, scale 1/8), then
// z[C,h] = sum_c p[c,C] * V[c,h] = P^T V.  K/Q/V^T are L2-resident (48 MB bf16
// total vs 192 MB L2), so fragments come straight from global.
__global__ __launch_bounds__(256) void attn_kernel(const bf16_t* __restrict__ Kb,
                                                   const bf16_t* __restrict__ Qb,
                                                   const bf16_t* __restrict__ VT,
                                                   bf16_t* __restrict__ Z) {
  __shared__ float  sm[SEQ];
  __shared__ float  sl[SEQ];
  __shared__ bf16_t sP[8][16 * 32];  // per-wave transposed probability tile

  const int bh = blockIdx.x;
  const int b  = bh >> 4;
  const int a  = bh & 15;
  const bf16_t* K = Kb + (size_t)bh * SEQ * DH;
  const bf16_t* Q = Qb + (size_t)bh * SEQ * DH;
  const bf16_t* V = VT + (size_t)bh * DH * SEQ;

  const int lane = threadIdx.x & 31;
  const int wave = threadIdx.x >> 5;
  const int hi   = lane >> 4;
  const int nn   = lane & 15;

  // ---- pass 1: per-row running max + softmax denominator ----
  for (int t = 0; t < 8; ++t) {
    const int r0 = t * 128 + wave * 16;
    v16bf aK0 = load_a(K, DH, r0, 0);
    v16bf aK1 = load_a(K, DH, r0, 32);
    float m8[8], l8[8];
#pragma unroll
    for (int e = 0; e < 8; ++e) { m8[e] = -3.0e38f; l8[e] = 0.0f; }

    for (int C0 = 0; C0 <= r0; C0 += 16) {  // causal: columns C <= c only
      v16bf bQ0 = load_b(Q, DH, C0, 0);
      v16bf bQ1 = load_b(Q, DH, C0, 32);
      v8f s = zero8();
      s = wmma_bf16(aK0, bQ0, s);
      s = wmma_bf16(aK1, bQ1, s);
#pragma unroll
      for (int e = 0; e < 8; ++e) {
        const int c = r0 + e + 8 * hi;
        const int C = C0 + nn;
        float sv = (C <= c) ? s[e] * 0.125f : -1.25e9f;  // (-1e10)/sqrt(64)
        float mt = sv;
#pragma unroll
        for (int off = 1; off < 16; off <<= 1) mt = fmaxf(mt, __shfl_xor(mt, off, 32));
        const float mn = fmaxf(m8[e], mt);
        float ex = __expf(sv - mn);
#pragma unroll
        for (int off = 1; off < 16; off <<= 1) ex += __shfl_xor(ex, off, 32);
        l8[e] = l8[e] * __expf(m8[e] - mn) + ex;
        m8[e] = mn;
      }
    }
    if (nn == 0) {
#pragma unroll
      for (int e = 0; e < 8; ++e) {
        sm[r0 + e + 8 * hi] = m8[e];
        sl[r0 + e + 8 * hi] = l8[e];
      }
    }
  }
  __syncthreads();

  // ---- pass 2: recompute scores, form P, z = P^T V ----
  for (int t = 0; t < 8; ++t) {
    const int C0 = t * 128 + wave * 16;
    v16bf bQ0 = load_b(Q, DH, C0, 0);
    v16bf bQ1 = load_b(Q, DH, C0, 32);
    v8f z0 = zero8(), z1 = zero8(), z2 = zero8(), z3 = zero8();
    bf16_t* P = &sP[wave][0];

    for (int c0 = (C0 & ~31); c0 < SEQ; c0 += 32) {
      v16bf aK0 = load_a(K, DH, c0, 0);
      v16bf aK1 = load_a(K, DH, c0, 32);
      v16bf aK2 = load_a(K, DH, c0 + 16, 0);
      v16bf aK3 = load_a(K, DH, c0 + 16, 32);
      v8f s0 = zero8(), s1 = zero8();
      s0 = wmma_bf16(aK0, bQ0, s0); s0 = wmma_bf16(aK1, bQ1, s0);
      s1 = wmma_bf16(aK2, bQ0, s1); s1 = wmma_bf16(aK3, bQ1, s1);

      // p = exp(s/8 - m)/l, stored TRANSPOSED into LDS: P[C_local][c_local]
#pragma unroll
      for (int half = 0; half < 2; ++half) {
        v8f sv = half ? s1 : s0;
#pragma unroll
        for (int e = 0; e < 8; ++e) {
          const int c = c0 + half * 16 + e + 8 * hi;
          const int C = C0 + nn;
          float p = 0.0f;
          if (C <= c) p = __expf(sv[e] * 0.125f - sm[c]) / sl[c];
          P[nn * 32 + half * 16 + e + 8 * hi] = (bf16_t)p;
        }
      }
      // Wave-local LDS RAW; DS ops are in-order per wave but keep the compiler
      // honest and show the gfx1250 split-counter wait.
      asm volatile("s_wait_dscnt 0" ::: "memory");

      v16bf aP  = load_a(P, 32, 0, 0);       // A = P^T tile: 16(C) x 32(c)
      v16bf bV0 = load_b(V, SEQ,  0, c0);    // B = V tile: 32(c) x 16(h), from V^T rows
      v16bf bV1 = load_b(V, SEQ, 16, c0);
      v16bf bV2 = load_b(V, SEQ, 32, c0);
      v16bf bV3 = load_b(V, SEQ, 48, c0);
      z0 = wmma_bf16(aP, bV0, z0);
      z1 = wmma_bf16(aP, bV1, z1);
      z2 = wmma_bf16(aP, bV2, z2);
      z3 = wmma_bf16(aP, bV3, z3);
    }

    // z_flat[b, C, a*64 + h]  (bf16 for the output GEMM)
#pragma unroll
    for (int e = 0; e < 8; ++e) {
      const size_t row = ((size_t)b * SEQ + (C0 + e + 8 * hi)) * EMB;
      const int col = a * DH + nn;
      Z[row + col +  0] = (bf16_t)z0[e];
      Z[row + col + 16] = (bf16_t)z1[e];
      Z[row + col + 32] = (bf16_t)z2[e];
      Z[row + col + 48] = (bf16_t)z3[e];
    }
  }
}

// ---------------------------------------------------------------------------
// Output projection: out[8192,1024] (f32) = Z @ W_O^T.
__global__ __launch_bounds__(256) void gemm_out_kernel(const bf16_t* __restrict__ Zb,
                                                       const bf16_t* __restrict__ Wo,
                                                       float* __restrict__ O) {
  __shared__ bf16_t sA[128 * 32];
  __shared__ bf16_t sB[128 * 32];
  const int lane = threadIdx.x & 31;
  const int wave = threadIdx.x >> 5;
  const int hi = lane >> 4, nn = lane & 15;
  const int rowB = blockIdx.x * 128;
  const int colB = blockIdx.y * 128;
  const int row0 = rowB + (wave & 3) * 32;
  const int col0 = colB + (wave >> 2) * 64;

  GemmAcc acc;
  gemm_block_bf16(Zb, Wo, sA, sB, rowB, colB, acc);

#pragma unroll
  for (int i = 0; i < 2; ++i)
#pragma unroll
    for (int j = 0; j < 4; ++j)
#pragma unroll
      for (int e = 0; e < 8; ++e) {
        const int rg = row0 + i * 16 + e + 8 * hi;
        const int cg = col0 + j * 16 + nn;
        O[(size_t)rg * EMB + cg] = acc.a[i][j][e];
      }
}

// ---------------------------------------------------------------------------
extern "C" void kernel_launch(void* const* d_in, const int* in_sizes, int n_in,
                              void* d_out, int out_size, void* d_ws, size_t ws_size,
                              hipStream_t stream) {
  (void)in_sizes; (void)n_in; (void)out_size; (void)ws_size;
  const float* x  = (const float*)d_in[0];
  const float* wk = (const float*)d_in[1];
  const float* wq = (const float*)d_in[2];
  const float* wv = (const float*)d_in[3];
  const float* wo = (const float*)d_in[4];

  // Workspace layout (bf16 elements), total ~88 MB — resident in 192 MB L2.
  bf16_t* p = (bf16_t*)d_ws;
  bf16_t* xb   = p; p += (size_t)ROWS * EMB;                 // x in bf16
  bf16_t* wqkv = p; p += (size_t)NQKV * EMB;                 // [K;Q;V] weight rows
  bf16_t* wob  = p; p += (size_t)EMB * EMB;                  // W_O rows
  bf16_t* Kb   = p; p += (size_t)BATCH * HEADS * SEQ * DH;   // [BH,S,64]
  bf16_t* Qb   = p; p += (size_t)BATCH * HEADS * SEQ * DH;   // [BH,S,64]
  bf16_t* VT   = p; p += (size_t)BATCH * HEADS * DH * SEQ;   // [BH,64,S]
  bf16_t* Zb   = p;                                          // z_flat [8192,1024]

  const int n_x = ROWS * EMB;          // 8388608
  const int n_w = HEADS * DH * EMB;    // 1048576 per projection
  cvt_f32_bf16<<<n_x / 256, 256, 0, stream>>>(x, xb, n_x);
  cvt_f32_bf16<<<n_w / 256, 256, 0, stream>>>(wk, wqkv,           n_w);
  cvt_f32_bf16<<<n_w / 256, 256, 0, stream>>>(wq, wqkv + n_w,     n_w);
  cvt_f32_bf16<<<n_w / 256, 256, 0, stream>>>(wv, wqkv + 2 * n_w, n_w);
  cvt_f32_bf16<<<n_w / 256, 256, 0, stream>>>(wo, wob,            EMB * EMB);

  gemm_qkv_kernel<<<dim3(ROWS / 128, NQKV / 128), 256, 0, stream>>>(xb, wqkv, Kb, Qb, VT);
  attn_kernel<<<BATCH * HEADS, 256, 0, stream>>>(Kb, Qb, VT, Zb);
  gemm_out_kernel<<<dim3(ROWS / 128, EMB / 128), 256, 0, stream>>>(Zb, wob, (float*)d_out);
}
